// DynamicRouting_57930518889134
// MI455X (gfx1250) — compile-verified, matrix operand order checked by reading, not run on hardware
//
#include <hip/hip_runtime.h>

#define C_DIM  256
#define HW     16384
#define NFEAT  4
#define TILE_P 16

typedef __bf16 v16bf __attribute__((ext_vector_type(16)));
typedef float  v8f   __attribute__((ext_vector_type(8)));

union Frag {
  v16bf v;
  __bf16 e[16];
  uint4 q[2];
};

union BfPair { __bf16 h[2]; unsigned int u; };

__device__ __forceinline__ void store_pair(__bf16* dst, float lo, float hi) {
  BfPair pr;
  pr.h[0] = (__bf16)lo;
  pr.h[1] = (__bf16)hi;
  *reinterpret_cast<unsigned int*>(dst) = pr.u;
}

// A fragment from pre-swizzled weights: one lane's 16 bf16 are contiguous (32B).
__device__ __forceinline__ Frag load_a_swz(const __bf16* base) {
  const uint4* p = reinterpret_cast<const uint4*>(base);
  Frag a;
  a.q[0] = p[0];
  a.q[1] = p[1];
  return a;
}

// B fragment (32x16 bf16): lane L holds col N=L%16; k-runs {0..7,16..23} (half 0)
// / {8..15,24..31} (half 1).  rowptr = &lds[col][kstep*32 + half*8].
__device__ __forceinline__ Frag load_b_frag(const __bf16* rowptr) {
  const uint4* bp = reinterpret_cast<const uint4*>(rowptr);
  Frag b;
  b.q[0] = bp[0];   // k-run 0 (8 bf16)
  b.q[1] = bp[2];   // k-run 1 starts +16 bf16
  return b;
}

__global__ void zero_k(float* __restrict__ p) { p[threadIdx.x] = 0.f; }

// Pre-convert fp32 weights to bf16 in WMMA-A fragment-ready layout:
//   wswz[(((n*16+mt)*8+ks)*32+lane)*16 + e]
// lane: m=lane&15 (row M), half=lane>>4; e<8 -> k=ks*32+half*8+e, e>=8 -> +16.
__global__ __launch_bounds__(256)
void swizzle_weights(const float* __restrict__ Wg, const float* __restrict__ outw,
                     __bf16* __restrict__ wswz, __bf16* __restrict__ oswz)
{
  const int idx = blockIdx.x * 256 + threadIdx.x;
  const int TOTAL_W = NFEAT * 16 * 8 * 32;        // 16384 fragment-lanes
  const int TOTAL_O = 16 * 8 * 32;                // 4096
  if (idx < TOTAL_W) {
    const int lane = idx & 31;
    const int ks   = (idx >> 5) & 7;
    const int mt   = (idx >> 8) & 15;
    const int n    = idx >> 12;
    const int m = lane & 15, half = lane >> 4;
    const int d = mt * 16 + m;
    const float* row = Wg + ((size_t)n * C_DIM + d) * C_DIM + ks * 32 + half * 8;
    Frag a;
    #pragma unroll
    for (int e2 = 0; e2 < 8; ++e2) a.e[e2]     = (__bf16)row[e2];
    #pragma unroll
    for (int e2 = 0; e2 < 8; ++e2) a.e[8 + e2] = (__bf16)row[16 + e2];
    uint4* dst = reinterpret_cast<uint4*>(wswz + (size_t)idx * 16);
    dst[0] = a.q[0]; dst[1] = a.q[1];
  } else if (idx < TOTAL_W + TOTAL_O) {
    const int j = idx - TOTAL_W;
    const int lane = j & 31;
    const int ks   = (j >> 5) & 7;
    const int mt   = j >> 8;
    const int m = lane & 15, half = lane >> 4;
    const int d = mt * 16 + m;
    const float* row = outw + (size_t)d * C_DIM + ks * 32 + half * 8;
    Frag a;
    #pragma unroll
    for (int e2 = 0; e2 < 8; ++e2) a.e[e2]     = (__bf16)row[e2];
    #pragma unroll
    for (int e2 = 0; e2 < 8; ++e2) a.e[8 + e2] = (__bf16)row[16 + e2];
    uint4* dst = reinterpret_cast<uint4*>(oswz + (size_t)j * 16);
    dst[0] = a.q[0]; dst[1] = a.q[1];
  }
}

__global__ __launch_bounds__(256)
void routing_main(const float* __restrict__ f0, const float* __restrict__ f1,
                  const float* __restrict__ f2, const float* __restrict__ f3,
                  const __bf16* __restrict__ wswz, const float* __restrict__ bg,
                  const __bf16* __restrict__ oswz, const float* __restrict__ outb,
                  float* __restrict__ pre, float* __restrict__ gsum,
                  float* __restrict__ gsq)
{
  __shared__ float tf[NFEAT][C_DIM][TILE_P];                     // 64 KB
  __shared__ __align__(16) __bf16 ldsB[NFEAT][TILE_P][C_DIM];    // 32 KB
  __shared__ float weighted[C_DIM][TILE_P];                      // 16 KB
  __shared__ __align__(16) __bf16 ldsWB[TILE_P][C_DIM];          //  8 KB
  __shared__ float bnsum[C_DIM], bnsq[C_DIM];
  __shared__ float part[256];                                    //  1 KB partials
  __shared__ float tfnorm[NFEAT][TILE_P], wnorm[TILE_P];
  __shared__ float logits[NFEAT][TILE_P], rw[NFEAT][TILE_P];

  const int t       = threadIdx.x;
  const int b       = blockIdx.y;
  const int posBase = blockIdx.x * TILE_P;
  const int lane    = t & 31;
  const int wave    = t >> 5;
  const int half    = lane >> 4;
  const int col     = lane & 15;

  bnsum[t] = 0.f; bnsq[t] = 0.f;

  // ---- Stage 1: features -> bf16 LDS [n][p][c], packed channel-pair b32 stores
  for (int r = t; r < NFEAT * 128 * 2; r += 256) {
    const int phalf = r & 1;                      // which 8 positions
    const int cp    = (r >> 1) & 127;             // channel pair
    const int n     = r >> 8;
    const int c0    = cp * 2;
    const int p0    = phalf * 8;
    const float* fn = (n == 0) ? f0 : (n == 1) ? f1 : (n == 2) ? f2 : f3;
    const float* base = fn + ((size_t)b * C_DIM + c0) * HW + posBase + p0;
    const float4* s0 = reinterpret_cast<const float4*>(base);
    const float4* s1 = reinterpret_cast<const float4*>(base + HW);
    float4 a0 = s0[0], a1 = s0[1];
    float4 b0 = s1[0], b1 = s1[1];
    store_pair(&ldsB[n][p0 + 0][c0], a0.x, b0.x);
    store_pair(&ldsB[n][p0 + 1][c0], a0.y, b0.y);
    store_pair(&ldsB[n][p0 + 2][c0], a0.z, b0.z);
    store_pair(&ldsB[n][p0 + 3][c0], a0.w, b0.w);
    store_pair(&ldsB[n][p0 + 4][c0], a1.x, b1.x);
    store_pair(&ldsB[n][p0 + 5][c0], a1.y, b1.y);
    store_pair(&ldsB[n][p0 + 6][c0], a1.z, b1.z);
    store_pair(&ldsB[n][p0 + 7][c0], a1.w, b1.w);
  }
  __syncthreads();

  // ---- Stage 2: tf[n] = W[n] @ feat[n] + b[n]   (64 WMMA tiles, 8 waves x 8)
  for (int T = wave; T < 64; T += 8) {
    const int n  = T >> 4;
    const int mt = T & 15;
    const __bf16* abase = wswz + ((size_t)((n * 16 + mt) * 8) * 32 + lane) * 16;
    v8f acc = {};
    #pragma unroll
    for (int ks = 0; ks < 8; ++ks) {
      if (ks < 7) __builtin_prefetch(abase + (size_t)(ks + 1) * 32 * 16, 0, 1);
      Frag a  = load_a_swz(abase + (size_t)ks * 32 * 16);
      Frag bf = load_b_frag(&ldsB[n][col][ks * 32 + half * 8]);
      acc = __builtin_amdgcn_wmma_f32_16x16x32_bf16(false, a.v, false, bf.v,
                                                    (short)0, acc, false, false);
    }
    const float4* bb = reinterpret_cast<const float4*>(bg + (size_t)n * C_DIM + mt * 16 + half * 8);
    float4 bv0 = bb[0], bv1 = bb[1];
    acc[0]+=bv0.x; acc[1]+=bv0.y; acc[2]+=bv0.z; acc[3]+=bv0.w;
    acc[4]+=bv1.x; acc[5]+=bv1.y; acc[6]+=bv1.z; acc[7]+=bv1.w;
    #pragma unroll
    for (int r = 0; r < 8; ++r)                   // D: VGPR r -> row r + 8*half
      tf[n][mt * 16 + r + 8 * half][col] = acc[r];
  }
  __syncthreads();

  // ---- Stage 3: routing entirely in LDS (reductions spread over all 256 thr)
  {                                               // ||tf||: 4 threads per (n,p)
    const int pr = t >> 2, sub = t & 3;
    const int n = pr >> 4, p = pr & 15;
    const int c0 = sub * 64;
    float s = 0.f;
    for (int c = c0; c < c0 + 64; ++c) { float v = tf[n][c][p]; s += v * v; }
    part[t] = s;
  }
  __syncthreads();
  if (t < 64) {
    const int n = t >> 4, p = t & 15;
    float s = part[t * 4] + part[t * 4 + 1] + part[t * 4 + 2] + part[t * 4 + 3];
    tfnorm[n][p] = fmaxf(sqrtf(s), 1e-8f);
    logits[n][p] = 0.f;
  }
  __syncthreads();

  for (int it = 0; it < 3; ++it) {
    if (t < 16) {                                 // softmax over n
      const int p = t;
      float l0=logits[0][p], l1=logits[1][p], l2=logits[2][p], l3=logits[3][p];
      float m  = fmaxf(fmaxf(l0,l1), fmaxf(l2,l3));
      float e0=__expf(l0-m), e1=__expf(l1-m), e2=__expf(l2-m), e3=__expf(l3-m);
      float inv = 1.f / (e0+e1+e2+e3);
      rw[0][p]=e0*inv; rw[1][p]=e1*inv; rw[2][p]=e2*inv; rw[3][p]=e3*inv;
    }
    __syncthreads();
    {                                             // weighted = sum_n tf * rw
      const int p = t & 15, cb = (t >> 4) << 4;
      #pragma unroll
      for (int j = 0; j < 16; ++j) {
        const int c = cb + j;
        weighted[c][p] = tf[0][c][p]*rw[0][p] + tf[1][c][p]*rw[1][p]
                       + tf[2][c][p]*rw[2][p] + tf[3][c][p]*rw[3][p];
      }
    }
    __syncthreads();
    if (it < 2) {
      {                                           // ||weighted||: 16 thr per p
        const int p = t >> 4, sub = t & 15;
        const int c0 = sub * 16;
        float s = 0.f;
        for (int c = c0; c < c0 + 16; ++c) { float v = weighted[c][p]; s += v * v; }
        part[t] = s;
      }
      __syncthreads();
      if (t < 16) {
        float s = 0.f;
        #pragma unroll
        for (int i = 0; i < 16; ++i) s += part[t * 16 + i];
        wnorm[t] = fmaxf(sqrtf(s), 1e-8f);
      }
      __syncthreads();
      {                                           // dot(tf, weighted): 4 per (n,p)
        const int pr = t >> 2, sub = t & 3;
        const int n = pr >> 4, p = pr & 15;
        const int c0 = sub * 64;
        float dot = 0.f;
        for (int c = c0; c < c0 + 64; ++c) dot += tf[n][c][p] * weighted[c][p];
        part[t] = dot;
      }
      __syncthreads();
      if (t < 64) {                               // cosine similarity update
        const int n = t >> 4, p = t & 15;
        float dot = part[t*4] + part[t*4+1] + part[t*4+2] + part[t*4+3];
        logits[n][p] += dot / (tfnorm[n][p] * wnorm[p]);
      }
      __syncthreads();
    }
  }

  {                                               // weighted -> bf16 [p][c]
    const int p = t & 15, cb = (t >> 4) << 4;
    #pragma unroll
    for (int j = 0; j < 16; j += 2)
      store_pair(&ldsWB[p][cb + j], weighted[cb + j][p], weighted[cb + j + 1][p]);
  }
  __syncthreads();

  // ---- Stage 4: output 1x1 conv via WMMA, pre-BN store + BN partial sums
  for (int mt = wave; mt < 16; mt += 8) {
    const __bf16* abase = oswz + ((size_t)(mt * 8) * 32 + lane) * 16;
    v8f acc = {};
    #pragma unroll
    for (int ks = 0; ks < 8; ++ks) {
      Frag a  = load_a_swz(abase + (size_t)ks * 32 * 16);
      Frag bf = load_b_frag(&ldsWB[col][ks * 32 + half * 8]);
      acc = __builtin_amdgcn_wmma_f32_16x16x32_bf16(false, a.v, false, bf.v,
                                                    (short)0, acc, false, false);
    }
    const float4* ob = reinterpret_cast<const float4*>(outb + mt * 16 + half * 8);
    float4 o0 = ob[0], o1 = ob[1];
    acc[0]+=o0.x; acc[1]+=o0.y; acc[2]+=o0.z; acc[3]+=o0.w;
    acc[4]+=o1.x; acc[5]+=o1.y; acc[6]+=o1.z; acc[7]+=o1.w;
    #pragma unroll
    for (int r = 0; r < 8; ++r) {
      const int dd = mt * 16 + r + 8 * half;
      const float v = acc[r];
      pre[((size_t)b * C_DIM + dd) * HW + posBase + col] = v;
      __hip_atomic_fetch_add(&bnsum[dd], v,     __ATOMIC_RELAXED, __HIP_MEMORY_SCOPE_WORKGROUP);
      __hip_atomic_fetch_add(&bnsq[dd],  v * v, __ATOMIC_RELAXED, __HIP_MEMORY_SCOPE_WORKGROUP);
    }
  }
  __syncthreads();
  __hip_atomic_fetch_add(&gsum[t], bnsum[t], __ATOMIC_RELAXED, __HIP_MEMORY_SCOPE_AGENT);
  __hip_atomic_fetch_add(&gsq[t],  bnsq[t],  __ATOMIC_RELAXED, __HIP_MEMORY_SCOPE_AGENT);
}

__global__ void bn_stats(const float* __restrict__ gsum, const float* __restrict__ gsq,
                         const float* __restrict__ gamma, const float* __restrict__ beta,
                         float* __restrict__ scale, float* __restrict__ shift)
{
  const int c = threadIdx.x;
  const float invN = 1.0f / (4.0f * (float)HW);
  float mean = gsum[c] * invN;
  float var  = gsq[c]  * invN - mean * mean;      // biased, as torch BN
  float sc   = gamma[c] * rsqrtf(var + 1e-5f);
  scale[c] = sc;
  shift[c] = beta[c] - mean * sc;
}

__global__ __launch_bounds__(256)
void bn_apply(const float4* __restrict__ pre4, const float* __restrict__ scale,
              const float* __restrict__ shift, float4* __restrict__ out4)
{
  const size_t i = (size_t)blockIdx.x * 256 + threadIdx.x;
  const int c = (int)((i * 4) >> 14) & 255;       // HW = 16384 per channel
  float4 v = pre4[i];
  const float sc = scale[c], sh = shift[c];
  v.x = fmaxf(v.x * sc + sh, 0.f);
  v.y = fmaxf(v.y * sc + sh, 0.f);
  v.z = fmaxf(v.z * sc + sh, 0.f);
  v.w = fmaxf(v.w * sc + sh, 0.f);
  out4[i] = v;
}

extern "C" void kernel_launch(void* const* d_in, const int* in_sizes, int n_in,
                              void* d_out, int out_size, void* d_ws, size_t ws_size,
                              hipStream_t stream) {
  const float* f0    = (const float*)d_in[0];
  const float* f1    = (const float*)d_in[1];
  const float* f2    = (const float*)d_in[2];
  const float* f3    = (const float*)d_in[3];
  const float* Wg    = (const float*)d_in[4];
  const float* bg    = (const float*)d_in[5];
  const float* outw  = (const float*)d_in[6];
  const float* outb  = (const float*)d_in[7];
  const float* gamma = (const float*)d_in[8];
  const float* beta  = (const float*)d_in[9];
  float* out = (float*)d_out;

  float*  pre   = (float*)d_ws;                   // 4*256*16384 fp32 = 64 MB
  float*  gsum  = pre + (size_t)4 * C_DIM * HW;
  float*  gsq   = gsum + C_DIM;
  float*  scale = gsq + C_DIM;
  float*  shift = scale + C_DIM;
  __bf16* wswz  = (__bf16*)(shift + C_DIM);       // 4*16*8*32*16 = 262144 bf16
  __bf16* oswz  = wswz + (size_t)NFEAT * 16 * 8 * 32 * 16;  // 65536 bf16

  zero_k<<<1, 512, 0, stream>>>(gsum);            // gsum + gsq

  swizzle_weights<<<(NFEAT * 16 * 8 * 32 + 16 * 8 * 32) / 256, 256, 0, stream>>>(
      Wg, outw, wswz, oswz);

  dim3 grid(HW / TILE_P, 4);                      // (1024, B)
  routing_main<<<grid, 256, 0, stream>>>(f0, f1, f2, f3, wswz, bg, oswz, outb,
                                         pre, gsum, gsq);

  bn_stats<<<1, C_DIM, 0, stream>>>(gsum, gsq, gamma, beta, scale, shift);

  bn_apply<<<(4 * C_DIM * HW) / 4 / 256, 256, 0, stream>>>(
      (const float4*)pre, scale, shift, (float4*)out);
}
